// Model_37589553774758
// MI455X (gfx1250) — matrix operand order repaired and checked by reading, not compiled, not run on hardware
//
#include <hip/hip_runtime.h>
#include <hip/hip_bf16.h>

typedef __bf16 bf16;
typedef __attribute__((ext_vector_type(16))) __bf16 bf16x16;
typedef __attribute__((ext_vector_type(8)))  __bf16 bf16x8;
typedef __attribute__((ext_vector_type(4)))  __bf16 bf16x4;
typedef __attribute__((ext_vector_type(8)))  float  f32x8;
typedef __attribute__((ext_vector_type(4)))  float  f32x4;

static constexpr int   N_NODES = 8192;
static constexpr int   D0 = 128, D1 = 256, D2 = 256, QH = 128;
static constexpr float EPSV = 1e-5f;

// ---------------------------------------------------------------- stats ----
__global__ void zero_stats_kernel(float* s) {  // 256 floats: sums[128], sumsq[128]
    s[threadIdx.x] = 0.f;
}

// 64 blocks x 256 threads; block b covers rows [128b, 128b+128)
__global__ void col_stats_kernel(const float* __restrict__ state,
                                 float* __restrict__ sums,
                                 float* __restrict__ sumsq) {
    const int t   = threadIdx.x;
    const int col = t & 127;
    const int rp  = t >> 7;
    const int r0  = blockIdx.x * 128 + rp;
    float s = 0.f, s2 = 0.f;
    #pragma unroll 4
    for (int i = 0; i < 64; ++i) {
        float v = state[(size_t)(r0 + 2 * i) * D0 + col];
        s += v; s2 += v * v;
    }
    __shared__ float sh[256], sh2[256];
    sh[t] = s; sh2[t] = s2;
    __syncthreads();
    if (t < 128) {
        atomicAdd(&sums[col],  sh[t]  + sh[t + 128]);
        atomicAdd(&sumsq[col], sh2[t] + sh2[t + 128]);
    }
}

// 1024 blocks x 256 threads x 4 elems = 8192*128
__global__ void normalize_bf16_kernel(const float* __restrict__ state,
                                      const float* __restrict__ sums,
                                      const float* __restrict__ sumsq,
                                      const float* __restrict__ gamma,
                                      const float* __restrict__ beta,
                                      bf16* __restrict__ xn) {
    const int idx = (blockIdx.x * 256 + threadIdx.x) * 4;
    f32x4 v = *(const f32x4*)(state + idx);
    bf16x4 o;
    #pragma unroll
    for (int j = 0; j < 4; ++j) {
        const int   c    = (idx + j) & (D0 - 1);
        const float mean = sums[c]  * (1.f / N_NODES);
        const float var  = sumsq[c] * (1.f / N_NODES) - mean * mean;
        const float r    = rsqrtf(var + EPSV);
        o[j] = (bf16)(((v[j] - mean) * r) * gamma[c] + beta[c]);
    }
    *(bf16x4*)(xn + idx) = o;
}

// ---------------------------------------------------------------- GEMM -----
// C = act(A[M,K] @ B + bias).
//   A: row-major; f32 (A_F32, kept f32 in LDS, cvt at fragment load) or bf16.
//      A tile staged with GLOBAL_LOAD_ASYNC_TO_LDS_B128 (ASYNCcnt tracked).
//   B: if B_F32: row-major f32 [K,BN], sync-staged + converted (small K only).
//      else:     bf16 stored TRANSPOSED in global as Bt[BN][K] (row stride K),
//                staged with async-to-LDS B128 copies.
//   C: bf16; TRANSC stores C^T[n][m] (ldc=M) with packed 16B stores.
// BM=64, BN=N, BK=64 (2 WMMA k-steps / stage), 256 thr = 8 waves (2x4),
// double-buffered LDS, one barrier per k-stage.
template <int BM, int BN, bool A_F32, bool B_F32, bool RELU, bool BIAS, bool TRANSC>
__global__ __launch_bounds__(256) void gemm_bf16_wmma(
        const void* __restrict__ Ap, const void* __restrict__ Bp,
        const float* __restrict__ bias, bf16* __restrict__ C, int K, int M) {
    constexpr int BK    = 64;
    constexpr int A_ESZ = A_F32 ? 4 : 2;
    constexpr int ASTRB = A_F32 ? (BK + 4) * 4 : (BK + 8) * 2;  // 272 / 144 B (16B-mult, odd*16)
    constexpr int ABUF  = BM * ASTRB;
    constexpr int BSTRB = (BK + 8) * 2;                          // 144 B
    constexpr int BBUF  = BN * BSTRB;
    constexpr int MT = 2;                 // m-tiles per wave (2 waves in M)
    constexpr int NT = BN / 64;           // n-tiles per wave (4 waves in N)
    constexpr int WC = BN / 4;
    constexpr int A_CPR = BK * A_ESZ / 16;        // 16B chunks per A row
    constexpr int A_NI  = BM * A_CPR / 256;       // async issues per thread (A)
    constexpr int B_NI  = BN * 8 / 256;           // async issues per thread (B bf16)

    __shared__ __align__(16) char smem[2 * ABUF + 2 * BBUF];
    char* const BsBase = smem + 2 * ABUF;

    const int t    = threadIdx.x;
    const int lane = t & 31;
    const int wid  = t >> 5;
    const int wm   = wid >> 2;
    const int wn   = wid & 3;
    const int ln16 = lane & 15;
    const int lh   = lane >> 4;
    const int blockM = blockIdx.x * BM;

    const unsigned as_base = (unsigned)(uintptr_t)smem;           // LDS byte offset
    const unsigned bs_base = as_base + 2u * ABUF;

    f32x8 acc[MT][NT];
    #pragma unroll
    for (int mt = 0; mt < MT; ++mt)
        #pragma unroll
        for (int nt = 0; nt < NT; ++nt)
            #pragma unroll
            for (int r = 0; r < 8; ++r) acc[mt][nt][r] = 0.f;

    // ---- tile staging (async A, async-or-convert B) ----
    auto stage = [&](int k0, int buf) {
        {   // A tile: BM x BK, raw async copy
            const unsigned long long abase = (unsigned long long)Ap;
            #pragma unroll
            for (int i = 0; i < A_NI; ++i) {
                const int chunk = i * 256 + t;
                const int row   = chunk / A_CPR;
                const int coff  = (chunk % A_CPR) * 16;
                const unsigned long long ga =
                    abase + ((unsigned long long)(blockM + row) * K + k0) * A_ESZ + coff;
                const unsigned la = as_base + (unsigned)(buf * ABUF + row * ASTRB + coff);
                asm volatile("global_load_async_to_lds_b128 %0, %1, off"
                             :: "v"(la), "v"(ga) : "memory");
            }
        }
        if constexpr (!B_F32) {   // Bt[n][k] bf16, raw async copy
            const unsigned long long bbase = (unsigned long long)Bp;
            #pragma unroll
            for (int i = 0; i < B_NI; ++i) {
                const int chunk = i * 256 + t;
                const int n     = chunk >> 3;
                const int coff  = (chunk & 7) * 16;
                const unsigned long long gb =
                    bbase + ((unsigned long long)n * K + k0) * 2 + coff;
                const unsigned lb = bs_base + (unsigned)(buf * BBUF + n * BSTRB + coff);
                asm volatile("global_load_async_to_lds_b128 %0, %1, off"
                             :: "v"(lb), "v"(gb) : "memory");
            }
        } else {                  // B row-major f32: sync convert + transpose-store
            const int bk  = t >> 2;                 // 64 k-rows, 4 thr/row
            const int bn0 = (t & 3) * (BN / 4);
            const float* src = (const float*)Bp + (size_t)(k0 + bk) * BN + bn0;
            char* dst = BsBase + buf * BBUF;
            #pragma unroll
            for (int j = 0; j < BN / 4; j += 4) {
                f32x4 q = *(const f32x4*)(src + j);
                #pragma unroll
                for (int u = 0; u < 4; ++u)
                    *(bf16*)(dst + (size_t)(bn0 + j + u) * BSTRB + bk * 2) = (bf16)q[u];
            }
        }
    };

    const int nk = K / BK;
    stage(0, 0);
    asm volatile("s_wait_asynccnt 0x0" ::: "memory");
    __syncthreads();

    for (int kt = 0; kt < nk; ++kt) {
        const int buf = kt & 1;
        if (kt + 1 < nk) stage((kt + 1) * BK, buf ^ 1);   // overlap with WMMAs below

        #pragma unroll
        for (int s = 0; s < 2; ++s) {                     // two WMMA k-steps per stage
            bf16x16 afr[MT];
            bf16x16 bfr[NT];
            #pragma unroll
            for (int mt = 0; mt < MT; ++mt) {
                const char* ap = smem + buf * ABUF +
                                 (wm * 32 + mt * 16 + ln16) * ASTRB +
                                 (s * 32 + lh * 8) * A_ESZ;
                if constexpr (A_F32) {
                    f32x4 a0 = *(const f32x4*)ap;
                    f32x4 a1 = *(const f32x4*)(ap + 16);
                    f32x4 a2 = *(const f32x4*)(ap + 64);   // +16 k elems
                    f32x4 a3 = *(const f32x4*)(ap + 80);
                    bf16x16 af;
                    #pragma unroll
                    for (int j = 0; j < 4; ++j) {
                        af[j]      = (bf16)a0[j];
                        af[4 + j]  = (bf16)a1[j];
                        af[8 + j]  = (bf16)a2[j];
                        af[12 + j] = (bf16)a3[j];
                    }
                    afr[mt] = af;
                } else {
                    bf16x8 lo = *(const bf16x8*)ap;
                    bf16x8 hi = *(const bf16x8*)(ap + 32); // +16 k elems
                    afr[mt] = __builtin_shufflevector(lo, hi, 0, 1, 2, 3, 4, 5, 6, 7,
                                                      8, 9, 10, 11, 12, 13, 14, 15);
                }
            }
            #pragma unroll
            for (int nt = 0; nt < NT; ++nt) {
                const char* bp = BsBase + buf * BBUF +
                                 (wn * WC + nt * 16 + ln16) * BSTRB +
                                 (s * 32 + lh * 8) * 2;
                bf16x8 lo = *(const bf16x8*)bp;
                bf16x8 hi = *(const bf16x8*)(bp + 32);
                bfr[nt] = __builtin_shufflevector(lo, hi, 0, 1, 2, 3, 4, 5, 6, 7,
                                                  8, 9, 10, 11, 12, 13, 14, 15);
            }
            #pragma unroll
            for (int mt = 0; mt < MT; ++mt)
                #pragma unroll
                for (int nt = 0; nt < NT; ++nt)
                    acc[mt][nt] = __builtin_amdgcn_wmma_f32_16x16x32_bf16(
                        false, afr[mt], false, bfr[nt], (short)0, acc[mt][nt], false, false);
        }

        asm volatile("s_wait_asynccnt 0x0" ::: "memory");
        __syncthreads();
    }

    // ---- epilogue ----
    #pragma unroll
    for (int mt = 0; mt < MT; ++mt) {
        #pragma unroll
        for (int nt = 0; nt < NT; ++nt) {
            const int n = wn * WC + nt * 16 + ln16;
            float bv = 0.f;
            if constexpr (BIAS) bv = bias[n];
            if constexpr (TRANSC) {
                bf16x8 o;
                #pragma unroll
                for (int r = 0; r < 8; ++r) {
                    float x = acc[mt][nt][r];
                    if constexpr (BIAS) x += bv;
                    if constexpr (RELU) x = fmaxf(x, 0.f);
                    o[r] = (bf16)x;
                }
                const int m0 = blockM + wm * 32 + mt * 16 + 8 * lh;
                *(bf16x8*)(C + (size_t)n * M + m0) = o;   // 16B packed store
            } else {
                #pragma unroll
                for (int r = 0; r < 8; ++r) {
                    const int m = blockM + wm * 32 + mt * 16 + r + 8 * lh;
                    float x = acc[mt][nt][r];
                    if constexpr (BIAS) x += bv;
                    if constexpr (RELU) x = fmaxf(x, 0.f);
                    C[(size_t)m * BN + n] = (bf16)x;
                }
            }
        }
    }
}

// ------------------------------------------------------------- Q head ------
__global__ void q_head_kernel(const bf16* __restrict__ P,
                              const float* __restrict__ Wq2,
                              const float* __restrict__ bq2,
                              float* __restrict__ out) {
    const int lane = threadIdx.x & 31;
    const int node = blockIdx.x * 8 + (threadIdx.x >> 5);
    bf16x4 pv = *(const bf16x4*)(P + (size_t)node * QH + lane * 4);
    f32x4  wv = *(const f32x4*)(Wq2 + lane * 4);
    float s = (float)pv[0] * wv[0] + (float)pv[1] * wv[1] +
              (float)pv[2] * wv[2] + (float)pv[3] * wv[3];
    #pragma unroll
    for (int off = 16; off; off >>= 1) s += __shfl_xor(s, off, 32);
    if (lane == 0) out[node] = s + bq2[0];
}

// ---------------------------------------------------------------- launch ---
extern "C" void kernel_launch(void* const* d_in, const int* in_sizes, int n_in,
                              void* d_out, int out_size, void* d_ws, size_t ws_size,
                              hipStream_t stream) {
    const float* state = (const float*)d_in[0];
    const float* A     = (const float*)d_in[1];
    const float* gamma = (const float*)d_in[2];
    const float* beta  = (const float*)d_in[3];
    const float* W1    = (const float*)d_in[4];
    const float* b1    = (const float*)d_in[5];
    const float* W2    = (const float*)d_in[6];
    const float* b2    = (const float*)d_in[7];
    const float* Wq1   = (const float*)d_in[8];
    const float* bq1   = (const float*)d_in[9];
    const float* Wq2   = (const float*)d_in[10];
    const float* bq2   = (const float*)d_in[11];
    float* out = (float*)d_out;

    // ws: stats(1KB) | Xn/P (2MB) | s1t (4MB, holds XW1^T then XW2^T) | s2 | s3
    char* ws    = (char*)d_ws;
    float* sums  = (float*)ws;
    float* sumsq = sums + 128;
    bf16* xn  = (bf16*)(ws + 1024);
    bf16* s1t = (bf16*)(ws + 1024 + 2u * 1024 * 1024);
    bf16* s2  = s1t + (size_t)N_NODES * D1;
    bf16* s3  = s2  + (size_t)N_NODES * D2;

    zero_stats_kernel<<<1, 256, 0, stream>>>(sums);
    col_stats_kernel<<<64, 256, 0, stream>>>(state, sums, sumsq);
    normalize_bf16_kernel<<<1024, 256, 0, stream>>>(state, sums, sumsq, gamma, beta, xn);

    const int gm = N_NODES / 64;  // 128 blocks
    // XW1^T[256,8192] = (Xn @ W1)^T
    gemm_bf16_wmma<64, 256, false, true, false, false, true>
        <<<gm, 256, 0, stream>>>(xn, W1, nullptr, s1t, D0, N_NODES);
    // H1 = relu(A @ XW1 + b1)   (B = XW1^T, async bf16 path)
    gemm_bf16_wmma<64, 256, true, false, true, true, false>
        <<<gm, 256, 0, stream>>>(A, s1t, b1, s2, N_NODES, N_NODES);
    // XW2^T = (H1 @ W2)^T
    gemm_bf16_wmma<64, 256, false, true, false, false, true>
        <<<gm, 256, 0, stream>>>(s2, W2, nullptr, s1t, D1, N_NODES);
    // H2 = relu(A @ XW2 + b2)
    gemm_bf16_wmma<64, 256, true, false, true, true, false>
        <<<gm, 256, 0, stream>>>(A, s1t, b2, s3, N_NODES, N_NODES);
    // P = relu(H2 @ Wq1 + bq1)  (reuse xn slot)
    gemm_bf16_wmma<64, 128, false, true, true, true, false>
        <<<gm, 256, 0, stream>>>(s3, Wq1, bq1, xn, D2, N_NODES);
    // q = P @ Wq2 + bq2
    q_head_kernel<<<1024, 256, 0, stream>>>(xn, Wq2, bq2, out);

    (void)in_sizes; (void)n_in; (void)out_size; (void)ws_size;
}
